// VisualTokenizer_18476949307916
// MI455X (gfx1250) — compile-verified, matrix operand order checked by reading, not run on hardware
//
#include <hip/hip_runtime.h>
#include <hip/hip_bf16.h>

// ---------------------------------------------------------------------------
// VisualTokenizer on MI455X (gfx1250, wave32, WMMA, async global->LDS)
//   1) pack patches -> bf16 A[18432,768]
//   2) pack conv_w  -> bf16 W[768,768]
//   3) pack codebook-> bf16 C[8192,768] + csq
//   4) GEMM1 (WMMA bf16): P = A*W^T + b -> bf16
//   5) GEMM2 (WMMA bf16) fused argmax of (P*C^T - 0.5*csq), packed u64 atomics
//   6) finalize: tokens (int32) + gather codebook rows (fp32)
// Mainloop: 256x128 block tile, 64x64 wave tile (16 WMMAs / 16 ds_load_b128
// per K-step), double-buffered LDS fed by GLOBAL_LOAD_ASYNC_TO_LDS_B128
// (ASYNCcnt), one barrier per K-step, peeled last iteration.
// ---------------------------------------------------------------------------

typedef __attribute__((ext_vector_type(16))) __bf16 v16bf;
typedef __attribute__((ext_vector_type(8)))  __bf16 v8bf;
typedef __attribute__((ext_vector_type(8)))  float  v8f;

#define BM 256
#define BN 128
#define BK 32
#define LDS_STRIDE 40   // halfs: 32 data + 8 pad = 80B rows -> conflict-free b128
#define THREADS 256     // 8 wave32 waves, 4 (M) x 2 (N)

union Frag { v16bf v; v8bf h[2]; };

// --- load one 16x32 bf16 A/B fragment from LDS (ISA 7.12.2 layout) ---------
__device__ __forceinline__ v16bf load_frag_lds(const __bf16* base, int row) {
  const int l  = threadIdx.x & 31;
  const int r  = row + (l & 15);
  const int kb = (l & 16) ? 8 : 0;
  const __bf16* p = base + r * LDS_STRIDE + kb;
  Frag f;
  f.h[0] = *(const v8bf*)(p);        // K = kb .. kb+7
  f.h[1] = *(const v8bf*)(p + 16);   // K = kb+16 .. kb+23
  return f.v;
}

// --- async copy: 256x32 bf16 A tile, one full row (4x16B) per thread -------
__device__ __forceinline__ void async_tile_a(const __bf16* __restrict__ g, int row0,
                                             int ld, int k0, __bf16* s, int tid) {
#pragma unroll
  for (int i = 0; i < 4; ++i) {
    int off = i * 8;                // halfs
    unsigned laddr = (unsigned)(uintptr_t)(s + tid * LDS_STRIDE + off);
    unsigned long long gaddr =
        (unsigned long long)(uintptr_t)(g + (size_t)(row0 + tid) * ld + k0 + off);
    asm volatile("global_load_async_to_lds_b128 %0, %1, off"
                 :: "v"(laddr), "v"(gaddr) : "memory");
  }
}

// --- async copy: 128x32 bf16 B tile, 2 chunks of 16B per thread ------------
__device__ __forceinline__ void async_tile_b(const __bf16* __restrict__ g, int row0,
                                             int ld, int k0, __bf16* s, int tid) {
#pragma unroll
  for (int i = 0; i < 2; ++i) {
    int c   = tid * 2 + i;          // 0..511
    int row = c >> 2;
    int off = (c & 3) * 8;          // halfs
    unsigned laddr = (unsigned)(uintptr_t)(s + row * LDS_STRIDE + off);
    unsigned long long gaddr =
        (unsigned long long)(uintptr_t)(g + (size_t)(row0 + row) * ld + k0 + off);
    asm volatile("global_load_async_to_lds_b128 %0, %1, off"
                 :: "v"(laddr), "v"(gaddr) : "memory");
  }
}

__device__ __forceinline__ void wait_async0() {
  asm volatile("s_wait_asynccnt 0x0" ::: "memory");
}

// --- one BK=32 step: 4 A frags x 4 B frags -> 16 WMMAs ---------------------
__device__ __forceinline__ void mma_step(const __bf16* As, const __bf16* Bs,
                                         int wy, int wx, v8f acc[4][4]) {
  v16bf a[4], b[4];
#pragma unroll
  for (int t = 0; t < 4; ++t) a[t] = load_frag_lds(As, wy * 64 + t * 16);
#pragma unroll
  for (int j = 0; j < 4; ++j) b[j] = load_frag_lds(Bs, wx * 64 + j * 16);
#pragma unroll
  for (int t = 0; t < 4; ++t)
#pragma unroll
    for (int j = 0; j < 4; ++j)
      acc[t][j] = __builtin_amdgcn_wmma_f32_16x16x32_bf16(
          false, a[t], false, b[j], (short)0, acc[t][j], false, false);
}

// --- shared double-buffered mainloop (peeled epilogue iteration) -----------
__device__ __forceinline__ void gemm_loop(const __bf16* __restrict__ A,
                                          const __bf16* __restrict__ B,
                                          int m0, int n0, int K, int tid,
                                          int wy, int wx,
                                          __bf16 (*As)[BM * LDS_STRIDE],
                                          __bf16 (*Bs)[BN * LDS_STRIDE],
                                          v8f acc[4][4]) {
  async_tile_a(A, m0, K, 0, As[0], tid);
  async_tile_b(B, n0, K, 0, Bs[0], tid);
  wait_async0();
  __syncthreads();
  const int nk = K / BK;
  for (int i = 0; i < nk - 1; ++i) {
    async_tile_a(A, m0, K, (i + 1) * BK, As[(i + 1) & 1], tid);
    async_tile_b(B, n0, K, (i + 1) * BK, Bs[(i + 1) & 1], tid);
    mma_step(As[i & 1], Bs[i & 1], wy, wx, acc);   // overlaps async copies
    wait_async0();
    __syncthreads();
  }
  mma_step(As[(nk - 1) & 1], Bs[(nk - 1) & 1], wy, wx, acc);
}

// --- GEMM1: P = A * B^T + bias, bf16 out -----------------------------------
__global__ __launch_bounds__(THREADS)
void vt_gemm_bias(const __bf16* __restrict__ A, const __bf16* __restrict__ B,
                  const float* __restrict__ bias, __bf16* __restrict__ P,
                  int K, int N) {
  __shared__ __bf16 As[2][BM * LDS_STRIDE];
  __shared__ __bf16 Bs[2][BN * LDS_STRIDE];
  const int tid  = threadIdx.x;
  const int wave = tid >> 5;
  const int wy = wave >> 1, wx = wave & 1;
  const int m0 = blockIdx.y * BM;
  const int n0 = blockIdx.x * BN;
  v8f acc[4][4] = {};
  gemm_loop(A, B, m0, n0, K, tid, wy, wx, As, Bs, acc);

  const int l = tid & 31;
  const int half = l >> 4;
  const int ncol = l & 15;
#pragma unroll
  for (int t = 0; t < 4; ++t)
#pragma unroll
    for (int j = 0; j < 4; ++j) {
      int col = n0 + wx * 64 + j * 16 + ncol;
      float bv = bias[col];
#pragma unroll
      for (int i = 0; i < 8; ++i) {
        int row = m0 + wy * 64 + t * 16 + half * 8 + i;
        P[(size_t)row * N + col] = (__bf16)(acc[t][j][i] + bv);
      }
    }
}

// --- GEMM2: fused argmax of (P*C^T - 0.5*csq) ------------------------------
__device__ __forceinline__ unsigned int f32_ord(float f) {
  unsigned int u = __float_as_uint(f);
  return (u & 0x80000000u) ? ~u : (u | 0x80000000u);
}

__global__ __launch_bounds__(THREADS)
void vt_gemm_argmax(const __bf16* __restrict__ P, const __bf16* __restrict__ C,
                    const float* __restrict__ csq,
                    unsigned long long* __restrict__ best, int K) {
  __shared__ __bf16 As[2][BM * LDS_STRIDE];
  __shared__ __bf16 Bs[2][BN * LDS_STRIDE];
  const int tid  = threadIdx.x;
  const int wave = tid >> 5;
  const int wy = wave >> 1, wx = wave & 1;
  const int m0 = blockIdx.y * BM;
  const int n0 = blockIdx.x * BN;
  v8f acc[4][4] = {};
  gemm_loop(P, C, m0, n0, K, tid, wy, wx, As, Bs, acc);

  const int l = tid & 31;
  const int half = l >> 4;
  const int ncol = l & 15;
#pragma unroll
  for (int t = 0; t < 4; ++t)
#pragma unroll
    for (int i = 0; i < 8; ++i) {
      unsigned long long pk = 0ull;
#pragma unroll
      for (int j = 0; j < 4; ++j) {
        int col = n0 + wx * 64 + j * 16 + ncol;
        float v = acc[t][j][i] - 0.5f * csq[col];
        unsigned long long cand =
            ((unsigned long long)f32_ord(v) << 32) | (unsigned int)(~col);
        pk = (cand > pk) ? cand : pk;
      }
      // reduce across the 16 lanes of each half (wave32)
#pragma unroll
      for (int s = 1; s < 16; s <<= 1) {
        unsigned long long o = __shfl_xor(pk, s, 32);
        pk = (o > pk) ? o : pk;
      }
      if (ncol == 0) {
        int row = m0 + wy * 64 + t * 16 + half * 8 + i;
        atomicMax(&best[row], pk);
      }
    }
}

// --- packing / prep kernels -------------------------------------------------
__global__ void vt_pack_patches(const float* __restrict__ img,
                                __bf16* __restrict__ Abf) {
  size_t idx = (size_t)blockIdx.x * blockDim.x + threadIdx.x; // < 18432*768
  int k = (int)(idx % 768);
  int m = (int)(idx / 768);
  int b = m / 576, pp = m % 576;
  int py = pp / 24, px = pp % 24;
  int c = k >> 8, r = k & 255, iy = r >> 4, ix = r & 15;
  float v = img[(((size_t)b * 3 + c) * 384 + (py * 16 + iy)) * 384 + (px * 16 + ix)];
  Abf[idx] = (__bf16)v;
}

__global__ void vt_pack_w(const float* __restrict__ w, __bf16* __restrict__ Wbf) {
  size_t idx = (size_t)blockIdx.x * blockDim.x + threadIdx.x; // < 768*768
  Wbf[idx] = (__bf16)w[idx];
}

__global__ __launch_bounds__(256)
void vt_pack_codebook(const float* __restrict__ cb, __bf16* __restrict__ Cbf,
                      float* __restrict__ csq) {
  __shared__ float red[256];
  const int row = blockIdx.x, tid = threadIdx.x;
  float s = 0.f;
#pragma unroll
  for (int i = 0; i < 3; ++i) {
    int col = tid + i * 256;
    float v = cb[(size_t)row * 768 + col];
    Cbf[(size_t)row * 768 + col] = (__bf16)v;
    s += v * v;
  }
  red[tid] = s;
  __syncthreads();
  for (int off = 128; off > 0; off >>= 1) {
    if (tid < off) red[tid] += red[tid + off];
    __syncthreads();
  }
  if (tid == 0) csq[row] = red[0];
}

__global__ void vt_init_best(unsigned long long* __restrict__ best, int n) {
  int i = blockIdx.x * blockDim.x + threadIdx.x;
  if (i < n) best[i] = 0ull;
}

__global__ __launch_bounds__(256)
void vt_finalize(const unsigned long long* __restrict__ best,
                 const float* __restrict__ cb, int* __restrict__ tok,
                 float* __restrict__ quant) {
  const int row = blockIdx.x, tid = threadIdx.x;
  unsigned int lo = (unsigned int)best[row];
  int col = (int)(~lo);               // stored as ~col
  if (tid == 0) tok[row] = col;
#pragma unroll
  for (int i = 0; i < 3; ++i) {
    int c = tid + i * 256;
    quant[(size_t)row * 768 + c] = cb[(size_t)col * 768 + c];
  }
}

// ---------------------------------------------------------------------------
extern "C" void kernel_launch(void* const* d_in, const int* in_sizes, int n_in,
                              void* d_out, int out_size, void* d_ws, size_t ws_size,
                              hipStream_t stream) {
  (void)in_sizes; (void)n_in; (void)out_size; (void)ws_size;
  const float* images   = (const float*)d_in[0];
  const float* conv_w   = (const float*)d_in[1];
  const float* conv_b   = (const float*)d_in[2];
  const float* codebook = (const float*)d_in[3];

  const int M = 32 * 24 * 24;   // 18432 patches
  const int D = 768;            // embed dim == K
  const int V = 8192;           // codebook size

  // workspace layout (all region sizes are multiples of 256B)
  char* ws = (char*)d_ws;
  __bf16* Abf = (__bf16*)ws;                 size_t o  = (size_t)M * D * 2;
  __bf16* Wbf = (__bf16*)(ws + o);           o += (size_t)D * D * 2;
  __bf16* Pbf = (__bf16*)(ws + o);           o += (size_t)M * D * 2;
  __bf16* Cbf = (__bf16*)(ws + o);           o += (size_t)V * D * 2;
  float*  csq = (float*)(ws + o);            o += (size_t)V * 4;
  unsigned long long* best = (unsigned long long*)(ws + o);

  vt_pack_patches<<<(M * D) / 256, 256, 0, stream>>>(images, Abf);
  vt_pack_w<<<(D * D) / 256, 256, 0, stream>>>(conv_w, Wbf);
  vt_pack_codebook<<<V, 256, 0, stream>>>(codebook, Cbf, csq);
  vt_init_best<<<(M + 255) / 256, 256, 0, stream>>>(best, M);

  vt_gemm_bias<<<dim3(D / BN, M / BM), THREADS, 0, stream>>>(
      Abf, Wbf, conv_b, Pbf, D, D);
  vt_gemm_argmax<<<dim3(V / BN, M / BM), THREADS, 0, stream>>>(
      Pbf, Cbf, csq, best, D);

  vt_finalize<<<M, 256, 0, stream>>>(best, codebook, (int*)d_out,
                                     (float*)d_out + M);
}
